// Decodeing_67164698575311
// MI455X (gfx1250) — matrix operand order repaired and checked by reading, not compiled, not run on hardware
//
#include <hip/hip_runtime.h>
#include <math.h>

#define B_      8
#define C_      32
#define H_      512
#define W_      256
#define HEAD_   64
#define NPK     18
#define RAD     5
#define THRESH_ 0.1f
#define EPS_    1e-5f

typedef __bf16 bf16;
typedef __bf16 v16bf __attribute__((ext_vector_type(16)));
typedef __bf16 v8bf  __attribute__((ext_vector_type(8)));
typedef float  v8f   __attribute__((ext_vector_type(8)));

// ---- d_out layout (flat float offsets, return order of reference tuple) ----
#define HM_SZ   (B_ * 2 * H_ * W_)           // dual_hm 2,097,152
#define HM_OFF  0
#define UP_OFF  (HM_SZ)                       // upper [8,18,2]
#define LO_OFF  (HM_SZ + B_ * NPK * 2)        // lower
#define MID_OFF (HM_SZ + 2 * B_ * NPK * 2)    // mid
#define VEC_OFF (HM_SZ + 3 * B_ * NPK * 2)    // vec_ind

// ---- d_ws layout (byte offsets); total use < 80KB ----
#define WS_PEAKS 0          // 16*18*2 floats = 2304B
#define WS_B     4096       // 72 B-fragments * 512 bf16 = 73728B (WMMA B layout)
#define WS_SS    45056      // scale[2*64] then shift[2*64] floats = 1024B
#define WS_W2    46080      // folded 1x1 weights [2 heads][2 oc][64] = 1024B
#define WS_B2    47104      // b2 [2 heads][2] floats

__device__ __forceinline__ int iabs_(int v) { return v < 0 ? -v : v; }

// ---------------------------------------------------------------------------
// Kernel 0: pack weights into WMMA fragment layouts + fold BN params.
// B-matrix (32x16 bf16) ISA layout: lanes 0-15 hold K=0..7,16..23 (2/VGPR), lanes
// 16-31 hold K=8..15,24..31, N = lane%16.  Fragment f=((h*9+t)*4+n), per-lane 16 bf16.
// ---------------------------------------------------------------------------
__global__ void prep_kernel(const float* __restrict__ w1_hm, const float* __restrict__ b1_hm,
                            const float* __restrict__ g_hm,  const float* __restrict__ be_hm,
                            const float* __restrict__ m_hm,  const float* __restrict__ v_hm,
                            const float* __restrict__ w2_hm, const float* __restrict__ b2_hm,
                            const float* __restrict__ w1_vec, const float* __restrict__ b1_vec,
                            const float* __restrict__ g_vec,  const float* __restrict__ be_vec,
                            const float* __restrict__ m_vec,  const float* __restrict__ v_vec,
                            const float* __restrict__ w2_vec, const float* __restrict__ b2_vec,
                            char* __restrict__ ws) {
    bf16*  wb  = (bf16*)(ws + WS_B);
    float* wss = (float*)(ws + WS_SS);
    float* ww2 = (float*)(ws + WS_W2);
    float* wb2 = (float*)(ws + WS_B2);
    int tid = threadIdx.x;

    for (int idx = tid; idx < 72 * 512; idx += blockDim.x) {
        int f = idx >> 9, within = idx & 511;
        int l = within >> 4, j = within & 15;
        int n = f & 3, ht = f >> 2;
        int t = ht % 9, h = ht / 9;
        int ic = (j & 7) + ((l >> 4) << 3) + ((j >> 3) << 4);       // K = input channel for tap t
        int oc = n * 16 + (l & 15);             // N = output channel
        const float* w1 = h ? w1_vec : w1_hm;   // [64][32][3][3]
        wb[idx] = (bf16)w1[(oc * C_ + ic) * 9 + t];
    }
    for (int idx = tid; idx < 128; idx += blockDim.x) {
        int h = idx >> 6, oc = idx & 63;
        const float* g  = h ? g_vec  : g_hm;
        const float* be = h ? be_vec : be_hm;
        const float* mm = h ? m_vec  : m_hm;
        const float* vv = h ? v_vec  : v_hm;
        const float* b1 = h ? b1_vec : b1_hm;
        float sc = g[oc] * rsqrtf(vv[oc] + EPS_);
        wss[idx]       = sc;                               // scale
        wss[128 + idx] = be[oc] + sc * (b1[oc] - mm[oc]);  // shift (folds conv bias)
    }
    for (int idx = tid; idx < 256; idx += blockDim.x) {
        int h = idx >> 7, r = idx & 127;                   // r = j*64+oc
        ww2[idx] = (h ? w2_vec : w2_hm)[r];
    }
    if (tid < 4) wb2[tid] = (tid < 2) ? b2_hm[tid] : b2_vec[tid - 2];
}

// ---------------------------------------------------------------------------
// Kernel 1: fused conv3x3(bf16 WMMA, f32 acc) + BN + ReLU + conv1x1 (+sigmoid)
// for BOTH heads.  Block = 256 thr = 8 waves; each wave owns 16 pixels of one
// row; block covers 128 contiguous pixels.  72 v_wmma per wave.
// ---------------------------------------------------------------------------
__global__ __launch_bounds__(256) void heads_kernel(const float* __restrict__ x,
                                                    const char* __restrict__ ws,
                                                    float* __restrict__ out) {
    __shared__ __align__(16) bf16  sX[3][136][32];   // rows y-1..y+1, cols x0-1..x0+128, ch innermost
    __shared__ __align__(16) float sH[8][16][64];    // per-wave ReLU'd hidden activations

    const bf16*  wb  = (const bf16*)(ws + WS_B);
    const float* wss = (const float*)(ws + WS_SS);
    const float* ww2 = (const float*)(ws + WS_W2);
    const float* wb2 = (const float*)(ws + WS_B2);

    int bi = blockIdx.x;
    int xb = bi & 1;
    int y  = (bi >> 1) & (H_ - 1);
    int b  = bi >> 10;
    int x0 = xb * 128;
    int tid = threadIdx.x;

    // stage input tile (coalesced along W per channel), zero-pad borders
    for (int idx = tid; idx < C_ * 3 * 130; idx += 256) {
        int c   = idx / 390;
        int rem = idx - c * 390;
        int ry  = rem / 130;
        int cx  = rem - ry * 130;
        int gy  = y - 1 + ry;
        int gx  = x0 - 1 + cx;
        float v = 0.f;
        if (gy >= 0 && gy < H_ && gx >= 0 && gx < W_)
            v = x[((b * C_ + c) * H_ + gy) * W_ + gx];
        sX[ry][cx][c] = (bf16)v;
    }
    __syncthreads();

    int wave = tid >> 5;
    int lane = tid & 31;
    int m    = lane & 15;                    // pixel row of A fragment
    int icb  = (lane < 16) ? 0 : 8;          // K-half per ISA A layout

    v8f acc[2][4] = {};                      // [head][n-tile], f32 accumulators

    #pragma unroll
    for (int t = 0; t < 9; ++t) {
        int ky = t / 3, kx = t - ky * 3;
        // A fragment 16x32 bf16: lanes<16 get K {0..7,16..23}, lanes>=16 {8..15,24..31}
        const bf16* ap = &sX[ky][wave * 16 + m + kx][icb];
        v8bf lo = *(const v8bf*)ap;          // ds_load_b128
        v8bf hi = *(const v8bf*)(ap + 16);   // ds_load_b128
        v16bf a = __builtin_shufflevector(lo, hi, 0,1,2,3,4,5,6,7,8,9,10,11,12,13,14,15);
        #pragma unroll
        for (int h = 0; h < 2; ++h) {
            #pragma unroll
            for (int n = 0; n < 4; ++n) {
                int f = (h * 9 + t) * 4 + n;
                v16bf bfr = *(const v16bf*)(wb + (f * 32 + lane) * 16);  // L2-resident
                acc[h][n] = __builtin_amdgcn_wmma_f32_16x16x32_bf16(
                    false, a, false, bfr, (short)0, acc[h][n], false, false);
            }
        }
    }

    // C layout: VGPR r -> (M=r, N=lane) for lanes 0-15, (M=8+r, N=lane-16) for 16-31
    int ncol  = lane & 15;
    int prow0 = (lane < 16) ? 0 : 8;
    int p2 = lane & 15;                      // pixel this lane finalizes
    int j  = lane >> 4;                      // output channel (0/1) this lane finalizes
    int gx = x0 + wave * 16 + p2;

    #pragma unroll
    for (int h = 0; h < 2; ++h) {
        #pragma unroll
        for (int n = 0; n < 4; ++n) {
            int oc = n * 16 + ncol;
            float sc = wss[h * 64 + oc];
            float sh = wss[128 + h * 64 + oc];
            #pragma unroll
            for (int r = 0; r < 8; ++r) {
                float hv = fmaxf(sc * acc[h][n][r] + sh, 0.f);   // BN + ReLU
                sH[wave][prow0 + r][oc] = hv;
            }
        }
        // wave-local LDS RAW: DS ops from the same wave complete in order.
        float s = wb2[h * 2 + j];
        const float4* hp = (const float4*)&sH[wave][p2][0];
        const float4* wp = (const float4*)&ww2[(h * 2 + j) * 64];
        #pragma unroll
        for (int q = 0; q < 16; ++q) {
            float4 hv = hp[q], wv = wp[q];
            s += hv.x * wv.x + hv.y * wv.y + hv.z * wv.z + hv.w * wv.w;
        }
        if (h == 0) {
            s = 1.0f / (1.0f + __expf(-s));                     // sigmoid
            out[HM_OFF + ((b * 2 + j) * H_ + y) * W_ + gx] = s;
        } else {
            out[VEC_OFF + ((b * 2 + j) * H_ + y) * W_ + gx] = s;
        }
    }
}

// ---------------------------------------------------------------------------
// Kernel 2: iterative argmax + square suppression per heatmap (16 maps).
// Suppressed elements read as 0 (all sigmoid values > 0 -> equivalent to the
// reference's in-place zeroing).  First-index tie-break matches jnp.argmax.
// ---------------------------------------------------------------------------
__global__ __launch_bounds__(1024) void peaks_kernel(const float* __restrict__ hm,
                                                     float* __restrict__ wsPeaks) {
    __shared__ float sV[1024];
    __shared__ int   sI[1024];
    __shared__ int   spx[NPK], spy[NPK], sval[NPK];
    int mapId = blockIdx.x;
    const float* map = hm + mapId * (H_ * W_);
    int tid = threadIdx.x;

    for (int it = 0; it < NPK; ++it) {
        float best = -1.0f;
        int   bidx = 0;
        for (int i = tid; i < H_ * W_; i += 1024) {
            int py = i >> 8;       // /W_ (W_=256)
            int px = i & 255;
            bool sup = false;
            for (int q = 0; q < it; ++q)
                if (sval[q] && iabs_(py - spy[q]) <= RAD && iabs_(px - spx[q]) <= RAD) sup = true;
            float v = sup ? 0.0f : map[i];
            if (v > best) { best = v; bidx = i; }   // ascending i -> first max wins
        }
        sV[tid] = best; sI[tid] = bidx;
        __syncthreads();
        for (int off = 512; off > 0; off >>= 1) {
            if (tid < off) {
                float v2 = sV[tid + off]; int i2 = sI[tid + off];
                if (v2 > sV[tid] || (v2 == sV[tid] && i2 < sI[tid])) { sV[tid] = v2; sI[tid] = i2; }
            }
            __syncthreads();
        }
        if (tid == 0) {
            int mi = sI[0];
            int py = mi >> 8, px = mi & 255;
            int valid = sV[0] > THRESH_;
            spx[it] = px; spy[it] = py; sval[it] = valid;
            wsPeaks[(mapId * NPK + it) * 2 + 0] = valid ? (float)px : 0.f;
            wsPeaks[(mapId * NPK + it) * 2 + 1] = valid ? (float)py : 0.f;
        }
        __syncthreads();
    }
}

// ---------------------------------------------------------------------------
// Kernel 3: stable sort by y descending (invalid -> -inf keys, zeros at end),
// then mid = (upper+lower)/2.
// ---------------------------------------------------------------------------
__global__ void order_kernel(const float* __restrict__ wsPeaks, float* __restrict__ out) {
    __shared__ float sPts[B_][2][NPK][2];
    int tid = threadIdx.x;
    if (tid < 16) {
        int b = tid >> 1, which = tid & 1;
        const float* src = wsPeaks + ((b * 2 + which) * NPK) * 2;
        float xs[NPK], ys[NPK], key[NPK];
        int valid[NPK];
        for (int i = 0; i < NPK; ++i) {
            xs[i] = src[2 * i]; ys[i] = src[2 * i + 1];
            valid[i] = (xs[i] + ys[i]) != 0.0f;
            key[i] = valid[i] ? ys[i] : -__builtin_inff();
        }
        for (int i = 0; i < NPK; ++i) {
            int rank = 0;
            for (int jj = 0; jj < NPK; ++jj)
                if (key[jj] > key[i] || (key[jj] == key[i] && jj < i)) rank++;
            float ox = valid[i] ? xs[i] : 0.f;
            float oy = valid[i] ? ys[i] : 0.f;
            sPts[b][which][rank][0] = ox;
            sPts[b][which][rank][1] = oy;
            int off = which ? LO_OFF : UP_OFF;
            out[off + (b * NPK + rank) * 2 + 0] = ox;
            out[off + (b * NPK + rank) * 2 + 1] = oy;
        }
    }
    __syncthreads();
    if (tid < 8) {
        int b = tid;
        for (int i = 0; i < NPK; ++i) {
            out[MID_OFF + (b * NPK + i) * 2 + 0] = 0.5f * (sPts[b][0][i][0] + sPts[b][1][i][0]);
            out[MID_OFF + (b * NPK + i) * 2 + 1] = 0.5f * (sPts[b][0][i][1] + sPts[b][1][i][1]);
        }
    }
}

extern "C" void kernel_launch(void* const* d_in, const int* in_sizes, int n_in,
                              void* d_out, int out_size, void* d_ws, size_t ws_size,
                              hipStream_t stream) {
    const float* x      = (const float*)d_in[0];
    const float* w1_hm  = (const float*)d_in[1];
    const float* b1_hm  = (const float*)d_in[2];
    const float* g_hm   = (const float*)d_in[3];
    const float* be_hm  = (const float*)d_in[4];
    const float* m_hm   = (const float*)d_in[5];
    const float* v_hm   = (const float*)d_in[6];
    const float* w2_hm  = (const float*)d_in[7];
    const float* b2_hm  = (const float*)d_in[8];
    const float* w1_ve  = (const float*)d_in[9];
    const float* b1_ve  = (const float*)d_in[10];
    const float* g_ve   = (const float*)d_in[11];
    const float* be_ve  = (const float*)d_in[12];
    const float* m_ve   = (const float*)d_in[13];
    const float* v_ve   = (const float*)d_in[14];
    const float* w2_ve  = (const float*)d_in[15];
    const float* b2_ve  = (const float*)d_in[16];

    char*  ws  = (char*)d_ws;
    float* out = (float*)d_out;

    prep_kernel<<<1, 256, 0, stream>>>(w1_hm, b1_hm, g_hm, be_hm, m_hm, v_hm, w2_hm, b2_hm,
                                       w1_ve, b1_ve, g_ve, be_ve, m_ve, v_ve, w2_ve, b2_ve, ws);
    heads_kernel<<<B_ * H_ * 2, 256, 0, stream>>>(x, ws, out);
    peaks_kernel<<<B_ * 2, 1024, 0, stream>>>(out + HM_OFF, (float*)(ws + WS_PEAKS));
    order_kernel<<<1, 32, 0, stream>>>((const float*)(ws + WS_PEAKS), out);
}
